// WRATBlock_81252191306589
// MI455X (gfx1250) — matrix operand order repaired and checked by reading, not compiled, not run on hardware
//
#include <hip/hip_runtime.h>
#include <cstdint>

#define D_MODEL 768
#define SEQ_L   1024
#define BATCH   8
#define NH      12
#define HEAD_D  64
#define D_FF    3072
#define ROWS    (BATCH * SEQ_L)   /* 8192 */
#define TAU_F   0.1f
#define LN_EPS  1e-5f

typedef unsigned short bf16_t;                                  // bf16 storage
typedef __bf16 v8bf  __attribute__((ext_vector_type(8)));
typedef __bf16 v16bf __attribute__((ext_vector_type(16)));
typedef float  v8f   __attribute__((ext_vector_type(8)));

// -------- async global->LDS (CDNA5). Guarded: falls back to sync copies ----
#if defined(__has_builtin)
#  if __has_builtin(__builtin_amdgcn_global_load_async_to_lds_b128)
#    define ASYNC_LDS 1
#  endif
#endif
#ifndef ASYNC_LDS
#  define ASYNC_LDS 0
#endif

#if ASYNC_LDS
typedef int v4i_t __attribute__((__vector_size__(16)));
typedef __attribute__((address_space(1))) v4i_t* gas_t;   // global v4i*
typedef __attribute__((address_space(3))) v4i_t* las_t;   // LDS v4i*
#endif

__device__ __forceinline__ void cp_b128(bf16_t* l, const bf16_t* g) {
#if ASYNC_LDS
  __builtin_amdgcn_global_load_async_to_lds_b128(
      (gas_t)(uintptr_t)g, (las_t)(unsigned)(uintptr_t)l, 0, 0);
#else
  *reinterpret_cast<uint4*>(l) = *reinterpret_cast<const uint4*>(g);
#endif
}
__device__ __forceinline__ void cp_wait() {
#if ASYNC_LDS
#  if __has_builtin(__builtin_amdgcn_s_wait_asynccnt)
  __builtin_amdgcn_s_wait_asynccnt(0);
#  else
  asm volatile("s_wait_asynccnt 0x0" ::: "memory");
#  endif
#endif
}

// ---------------------------------------------------------------- helpers ---
__device__ __forceinline__ bf16_t f2bf(float f) {               // RNE f32->bf16
  unsigned u = __float_as_uint(f);
  return (bf16_t)((u + 0x7FFFu + ((u >> 16) & 1u)) >> 16);
}
__device__ __forceinline__ v8f v8f_zero() {
  v8f z = {0.f, 0.f, 0.f, 0.f, 0.f, 0.f, 0.f, 0.f};
  return z;
}
// A-fragment (16x32 bf16): lane half kb=hi*8 holds K {kb..kb+7, kb+16..kb+23}
__device__ __forceinline__ v16bf frag_a(const bf16_t* p0, const bf16_t* p1) {
  v8bf lo = *reinterpret_cast<const v8bf*>(p0);
  v8bf hi = *reinterpret_cast<const v8bf*>(p1);
  return __builtin_shufflevector(lo, hi, 0,1,2,3,4,5,6,7,8,9,10,11,12,13,14,15);
}
// B-fragment (32x16 bf16): lane holds col N=lane&15, K {hi*8..hi*8+7, hi*8+16..hi*8+23}
__device__ __forceinline__ v16bf frag_b(const bf16_t* p) {
  v8bf lo = *reinterpret_cast<const v8bf*>(p);
  v8bf hi = *reinterpret_cast<const v8bf*>(p + 16);
  return __builtin_shufflevector(lo, hi, 0,1,2,3,4,5,6,7,8,9,10,11,12,13,14,15);
}
__device__ __forceinline__ v8f wmma_bf16(v16bf a, v16bf b, v8f c) {
  return __builtin_amdgcn_wmma_f32_16x16x32_bf16(false, a, false, b, (short)0, c,
                                                 false, false);
}

// --------------- weight convert + transpose: W[K,N] f32 -> Wt[N,K] bf16 ----
__global__ __launch_bounds__(256) void conv_t_w_kernel(
    const float* __restrict__ w, bf16_t* __restrict__ wt, int K, int N) {
  __shared__ float tile[32][33];
  const int k0 = blockIdx.x * 32, n0 = blockIdx.y * 32;
  const int tx = threadIdx.x, ty = threadIdx.y;             // block (32,8)
  for (int i = ty; i < 32; i += 8)
    tile[i][tx] = w[(size_t)(k0 + i) * N + n0 + tx];        // tile[k][n]
  __syncthreads();
  for (int i = ty; i < 32; i += 8)
    wt[(size_t)(n0 + i) * K + k0 + tx] = f2bf(tile[tx][i]);
}

// --------------------------------------- [B,D,L] -> [B,L,D] (f32 + bf16) ---
__global__ __launch_bounds__(256) void transpose_in_kernel(
    const float* __restrict__ in, float* __restrict__ outF,
    bf16_t* __restrict__ outB) {
  __shared__ float tile[32][33];
  const int l0 = blockIdx.x * 32, d0 = blockIdx.y * 32, b = blockIdx.z;
  const int tx = threadIdx.x, ty = threadIdx.y;
  for (int i = ty; i < 32; i += 8)
    tile[i][tx] = in[((size_t)b * D_MODEL + d0 + i) * SEQ_L + l0 + tx];
  __syncthreads();
  for (int i = ty; i < 32; i += 8) {
    float v = tile[tx][i];
    size_t idx = ((size_t)b * SEQ_L + l0 + i) * D_MODEL + d0 + tx;
    outF[idx] = v;
    outB[idx] = f2bf(v);
  }
}

// ------------------------------- out[b,d,l] = a[b,l,d] + c[b,l,d] (final) ---
__global__ __launch_bounds__(256) void add_transpose_out_kernel(
    const float* __restrict__ a, const float* __restrict__ c,
    float* __restrict__ out) {
  __shared__ float tile[32][33];
  const int l0 = blockIdx.x * 32, d0 = blockIdx.y * 32, b = blockIdx.z;
  const int tx = threadIdx.x, ty = threadIdx.y;
  for (int i = ty; i < 32; i += 8) {
    size_t idx = ((size_t)b * SEQ_L + l0 + i) * D_MODEL + d0 + tx;
    tile[i][tx] = a[idx] + c[idx];
  }
  __syncthreads();
  for (int i = ty; i < 32; i += 8)
    out[((size_t)b * D_MODEL + d0 + i) * SEQ_L + l0 + tx] = tile[tx][i];
}

// --------------------------------- energy[b,l] = mean_d |seq[b,l,d]| --------
__global__ __launch_bounds__(256) void energy_kernel(
    const float* __restrict__ seq, float* __restrict__ en) {
  const int lane = threadIdx.x & 31, wid = threadIdx.x >> 5;
  const int row = blockIdx.x * 8 + wid;
  const float* p = seq + (size_t)row * D_MODEL;
  float s = 0.f;
  #pragma unroll
  for (int i = 0; i < D_MODEL / 32; ++i) s += fabsf(p[lane + i * 32]);
  #pragma unroll
  for (int m = 1; m < 32; m <<= 1) s += __shfl_xor(s, m, 32);
  if (lane == 0) en[row] = s * (1.0f / D_MODEL);
}

// ----------------------------------------------------------- WMMA GEMM -----
// C[M,N] = A[M,K](bf16) x Wt[N,K](bf16, pre-transposed) + bias.
// Block 256 thr = 8 waves; tile 128x128; waves 2(M) x 4(N); wave tile 64x32
// (4x2 subtiles => 8 v_wmma / wave / K-step). Double-buffered async staging.
template <int GELU, bool HASF, bool HASB, bool VT>
__global__ __launch_bounds__(256) void gemm_bf16_kernel(
    const bf16_t* __restrict__ A, const bf16_t* __restrict__ Wt,
    const float* __restrict__ bias, float* __restrict__ Cf,
    bf16_t* __restrict__ Cb, bf16_t* __restrict__ CbT,
    int M, int N, int K) {
  __shared__ __align__(16) bf16_t As[2][128][32];   // [m][k]
  __shared__ __align__(16) bf16_t Bs[2][128][32];   // [n][k]
  const int tid  = threadIdx.x;
  const int lane = tid & 31, wid = tid >> 5;
  const int wm   = wid & 1, wn = wid >> 1;
  const int l15  = lane & 15, hib = lane >> 4;
  const int kb   = hib * 8;
  const int m0   = blockIdx.x * 128, n0 = blockIdx.y * 128;
  const int srow = tid >> 1, sch = tid & 1;         // staging: 32B per thread

  v8f acc[4][2];
  #pragma unroll
  for (int i = 0; i < 4; ++i)
    #pragma unroll
    for (int j = 0; j < 2; ++j) acc[i][j] = v8f_zero();

  auto stage = [&](int buf, int k0) {
    const bf16_t* asrc = &A [(size_t)(m0 + srow) * K + k0 + sch * 16];
    cp_b128(&As[buf][srow][sch * 16],     asrc);
    cp_b128(&As[buf][srow][sch * 16 + 8], asrc + 8);
    const bf16_t* bsrc = &Wt[(size_t)(n0 + srow) * K + k0 + sch * 16];
    cp_b128(&Bs[buf][srow][sch * 16],     bsrc);
    cp_b128(&Bs[buf][srow][sch * 16 + 8], bsrc + 8);
  };

  stage(0, 0);
  cp_wait();
  __syncthreads();
  const int nk = K / 32;
  for (int kt = 0; kt < nk; ++kt) {
    const int buf = kt & 1;
    if (kt + 1 < nk) stage(buf ^ 1, (kt + 1) * 32);  // prefetch next tile
    v16bf af[4], bfv[2];
    #pragma unroll
    for (int mi = 0; mi < 4; ++mi) {
      int row = wm * 64 + mi * 16 + l15;
      af[mi] = frag_a(&As[buf][row][kb], &As[buf][row][kb + 16]);
    }
    #pragma unroll
    for (int ni = 0; ni < 2; ++ni) {
      int col = wn * 32 + ni * 16 + l15;
      bfv[ni] = frag_b(&Bs[buf][col][hib * 8]);
    }
    #pragma unroll
    for (int mi = 0; mi < 4; ++mi)
      #pragma unroll
      for (int ni = 0; ni < 2; ++ni)
        acc[mi][ni] = wmma_bf16(af[mi], bfv[ni], acc[mi][ni]);
    cp_wait();
    __syncthreads();
  }
  // epilogue: C/D layout row = r + 8*hib, col = l15 (all modes compile-time)
  #pragma unroll
  for (int mi = 0; mi < 4; ++mi)
    #pragma unroll
    for (int ni = 0; ni < 2; ++ni) {
      const int n = n0 + wn * 32 + ni * 16 + l15;
      const float bv = bias[n];
      #pragma unroll
      for (int r = 0; r < 8; ++r) {
        const int m = m0 + wm * 64 + mi * 16 + r + hib * 8;
        float v = acc[mi][ni][r] + bv;
        if (GELU) v = 0.5f * v * (1.0f + erff(v * 0.70710678118f));
        if (HASF) Cf[(size_t)m * N + n] = v;
        if (HASB) Cb[(size_t)m * N + n] = f2bf(v);
        if (VT)   // per-head transposed store: [b, h, dh][L] for attention V
          CbT[(((size_t)(m >> 10) * NH + (n >> 6)) * HEAD_D + (n & 63)) * SEQ_L +
              (m & (SEQ_L - 1))] = f2bf(v);
      }
    }
}

// ----------------------------------------------------- flash attention -----
// grid (L/64, H, B), 128 thr = 4 waves; wave owns 16 q rows; 32-key tiles,
// double-buffered async K/V staging (V pre-transposed per head: [b,h,d][L]).
template <bool MASKED>
__global__ __launch_bounds__(128) void flash_attn_kernel(
    const bf16_t* __restrict__ Qb, const bf16_t* __restrict__ Kb,
    const bf16_t* __restrict__ Vtg, const float* __restrict__ energy, float tau,
    bf16_t* __restrict__ Ob) {
  __shared__ __align__(16) bf16_t Ks[2][32][64];   // [key][d] == B^T for QK^T
  __shared__ __align__(16) bf16_t Vt[2][64][32];   // [d][key] == B^T for P*V
  __shared__ __align__(16) bf16_t Ps[4][16][32];   // per-wave P restage
  const int tid  = threadIdx.x;
  const int lane = tid & 31, w = tid >> 5;
  const int l15  = lane & 15, hib = lane >> 4;
  const int kb   = hib * 8;
  const int b    = blockIdx.z, h = blockIdx.y;
  const int q0   = blockIdx.x * 64 + w * 16;
  const int krow = tid >> 2, kch = tid & 3;        // K staging: 32 rows x 4
  const int vrow = tid >> 1, vch = tid & 1;        // V staging: 64 rows x 2

  const bf16_t* qrow = Qb + ((size_t)(b * SEQ_L + q0 + l15) * D_MODEL + h * HEAD_D);
  const v16bf qf0 = frag_a(qrow + kb,      qrow + kb + 16);
  const v16bf qf1 = frag_a(qrow + 32 + kb, qrow + 32 + kb + 16);

  v8f o[4];
  float mi[8], li[8];
  #pragma unroll
  for (int i = 0; i < 4; ++i) o[i] = v8f_zero();
  #pragma unroll
  for (int r = 0; r < 8; ++r) { mi[r] = -__builtin_inff(); li[r] = 0.f; }
  const float scale = 0.125f;  // 1/sqrt(64)
  const float NEG = -__builtin_inff();

  auto stage = [&](int buf, int kt0) {
    const bf16_t* ksrc =
        Kb + ((size_t)(b * SEQ_L + kt0 + krow) * D_MODEL + h * HEAD_D + kch * 16);
    cp_b128(&Ks[buf][krow][kch * 16],     ksrc);
    cp_b128(&Ks[buf][krow][kch * 16 + 8], ksrc + 8);
    const bf16_t* vsrc =
        Vtg + ((size_t)(b * NH + h) * HEAD_D + vrow) * SEQ_L + kt0 + vch * 16;
    cp_b128(&Vt[buf][vrow][vch * 16],     vsrc);
    cp_b128(&Vt[buf][vrow][vch * 16 + 8], vsrc + 8);
  };

  stage(0, 0);
  cp_wait();
  __syncthreads();
  for (int kt = 0; kt < SEQ_L / 32; ++kt) {
    const int buf = kt & 1;
    const int kt0 = kt * 32;
    if (kt0 + 32 < SEQ_L) stage(buf ^ 1, kt0 + 32);   // prefetch next tile

    // S[16q x 32k] = Q K^T : 4 wmma
    v8f s0 = v8f_zero(), s1 = v8f_zero();
    s0 = wmma_bf16(qf0, frag_b(&Ks[buf][l15][hib * 8]),           s0);
    s0 = wmma_bf16(qf1, frag_b(&Ks[buf][l15][32 + hib * 8]),      s0);
    s1 = wmma_bf16(qf0, frag_b(&Ks[buf][16 + l15][hib * 8]),      s1);
    s1 = wmma_bf16(qf1, frag_b(&Ks[buf][16 + l15][32 + hib * 8]), s1);

    bool keep0 = true, keep1 = true;
    if (MASKED) {
      keep0 = energy[(size_t)b * SEQ_L + kt0 + l15] > tau;
      keep1 = energy[(size_t)b * SEQ_L + kt0 + 16 + l15] > tau;
    }
    // online softmax (rows r+8*hib; shfl masks <16 keep halves independent)
    #pragma unroll
    for (int r = 0; r < 8; ++r) {
      float a0 = keep0 ? s0[r] * scale : NEG;
      float a1 = keep1 ? s1[r] * scale : NEG;
      float rm = fmaxf(a0, a1);
      #pragma unroll
      for (int m = 1; m < 16; m <<= 1) rm = fmaxf(rm, __shfl_xor(rm, m, 32));
      float mnew = fmaxf(mi[r], rm);
      float alpha, p0, p1;
      if (mnew == NEG) { alpha = 1.f; p0 = 0.f; p1 = 0.f; }  // fully masked
      else {
        alpha = (mi[r] == NEG) ? 0.f : __expf(mi[r] - mnew);
        p0 = (a0 == NEG) ? 0.f : __expf(a0 - mnew);
        p1 = (a1 == NEG) ? 0.f : __expf(a1 - mnew);
      }
      float ps = p0 + p1;
      #pragma unroll
      for (int m = 1; m < 16; m <<= 1) ps += __shfl_xor(ps, m, 32);
      li[r] = li[r] * alpha + ps;
      mi[r] = mnew;
      #pragma unroll
      for (int nd = 0; nd < 4; ++nd) o[nd][r] *= alpha;
      Ps[w][r + hib * 8][l15]      = f2bf(p0);
      Ps[w][r + hib * 8][16 + l15] = f2bf(p1);
    }
    // O[16q x 64d] += P V : 4 wmma (same-wave LDS RAW is in-order)
    {
      const bf16_t* prow = &Ps[w][l15][0];
      v16bf pf = frag_a(prow + kb, prow + kb + 16);
      #pragma unroll
      for (int nd = 0; nd < 4; ++nd) {
        v16bf vf = frag_b(&Vt[buf][nd * 16 + l15][hib * 8]);
        o[nd] = wmma_bf16(pf, vf, o[nd]);
      }
    }
    cp_wait();
    __syncthreads();
  }
  float rcp[8];
  #pragma unroll
  for (int r = 0; r < 8; ++r) rcp[r] = li[r] > 0.f ? 1.f / li[r] : 0.f;
  #pragma unroll
  for (int nd = 0; nd < 4; ++nd) {
    const int d = nd * 16 + l15;
    #pragma unroll
    for (int r = 0; r < 8; ++r) {
      const int m = q0 + r + hib * 8;
      Ob[(size_t)(b * SEQ_L + m) * D_MODEL + h * HEAD_D + d] = f2bf(o[nd][r] * rcp[r]);
    }
  }
}

// -------------------------------------------- residual add + layernorm -----
template <bool HASA2>
__global__ __launch_bounds__(256) void add_ln_kernel(
    const float* __restrict__ x, const float* __restrict__ a1,
    const float* __restrict__ a2, const float* __restrict__ g,
    const float* __restrict__ bb, float* __restrict__ outF,
    bf16_t* __restrict__ outB) {
  const int lane = threadIdx.x & 31, wid = threadIdx.x >> 5;
  const int row = blockIdx.x * 8 + wid;
  const size_t base = (size_t)row * D_MODEL;
  float v[D_MODEL / 32];
  float s = 0.f;
  #pragma unroll
  for (int i = 0; i < D_MODEL / 32; ++i) {
    int d = lane + i * 32;
    float t = x[base + d] + a1[base + d];
    if (HASA2) t += a2[base + d];
    v[i] = t; s += t;
  }
  #pragma unroll
  for (int m = 1; m < 32; m <<= 1) s += __shfl_xor(s, m, 32);
  float mu = s * (1.f / D_MODEL);
  float var = 0.f;
  #pragma unroll
  for (int i = 0; i < D_MODEL / 32; ++i) { float d0 = v[i] - mu; var += d0 * d0; }
  #pragma unroll
  for (int m = 1; m < 32; m <<= 1) var += __shfl_xor(var, m, 32);
  float rstd = rsqrtf(var * (1.f / D_MODEL) + LN_EPS);
  #pragma unroll
  for (int i = 0; i < D_MODEL / 32; ++i) {
    int d = lane + i * 32;
    float y = (v[i] - mu) * rstd * g[d] + bb[d];
    outF[base + d] = y;
    outB[base + d] = f2bf(y);
  }
}

// =========================================================== host driver ===
extern "C" void kernel_launch(void* const* d_in, const int* in_sizes, int n_in,
                              void* d_out, int out_size, void* d_ws,
                              size_t ws_size, hipStream_t stream) {
  (void)in_sizes; (void)n_in; (void)out_size; (void)ws_size;
  const float* LL = (const float*)d_in[0];
  const float* LH = (const float*)d_in[1];
  struct Attn { const float *wq,*bq,*wk,*bk,*wv,*bv,*wo,*bo; };
  auto getAttn = [&](int i) {
    Attn a;
    a.wq = (const float*)d_in[i];     a.bq = (const float*)d_in[i + 1];
    a.wk = (const float*)d_in[i + 2]; a.bk = (const float*)d_in[i + 3];
    a.wv = (const float*)d_in[i + 4]; a.bv = (const float*)d_in[i + 5];
    a.wo = (const float*)d_in[i + 6]; a.bo = (const float*)d_in[i + 7];
    return a;
  };
  Attn aLL = getAttn(2), aLH = getAttn(10), aX = getAttn(18);
  const float* mLL_w1 = (const float*)d_in[26]; const float* mLL_b1 = (const float*)d_in[27];
  const float* mLL_w2 = (const float*)d_in[28]; const float* mLL_b2 = (const float*)d_in[29];
  const float* mLH_w1 = (const float*)d_in[30]; const float* mLH_b1 = (const float*)d_in[31];
  const float* mLH_w2 = (const float*)d_in[32]; const float* mLH_b2 = (const float*)d_in[33];
  const float* g1 = (const float*)d_in[34]; const float* b1n = (const float*)d_in[35];
  const float* g2 = (const float*)d_in[36]; const float* b2n = (const float*)d_in[37];

  char* wsb = (char*)d_ws;
  size_t cur = 0;
  auto alloc = [&](size_t bytes) -> void* {
    void* p = wsb + cur;
    cur += (bytes + 255) & ~(size_t)255;
    return p;
  };
  const size_t NE = (size_t)ROWS * D_MODEL;
  float*  seqLL_F = (float*)alloc(NE * 4);
  float*  seqLH_F = (float*)alloc(NE * 4);
  bf16_t* seqLL_B = (bf16_t*)alloc(NE * 2);
  bf16_t* seqLH_B = (bf16_t*)alloc(NE * 2);
  float*  energy  = (float*)alloc((size_t)ROWS * 4);
  bf16_t* qB      = (bf16_t*)alloc(NE * 2);
  bf16_t* kB      = (bf16_t*)alloc(NE * 2);
  bf16_t* vTB     = (bf16_t*)alloc(NE * 2);      // per-head transposed V
  bf16_t* attnB   = (bf16_t*)alloc(NE * 2);
  float*  outF_LL = (float*)alloc(NE * 4);
  float*  outF_LH = (float*)alloc(NE * 4);
  float*  outF_X  = (float*)alloc(NE * 4);
  bf16_t* outB_LL = (bf16_t*)alloc(NE * 2);
  bf16_t* outB_LH = (bf16_t*)alloc(NE * 2);
  float*  fusLL_F = (float*)alloc(NE * 4);
  float*  fusLH_F = (float*)alloc(NE * 4);
  bf16_t* fusLL_B = (bf16_t*)alloc(NE * 2);
  bf16_t* fusLH_B = (bf16_t*)alloc(NE * 2);
  bf16_t* hiddenB = (bf16_t*)alloc((size_t)ROWS * D_FF * 2);
  float*  mlpF    = (float*)alloc(NE * 4);

  // bf16 transposed weight pool Wt[N][K] (rebuilt every call: deterministic)
  const float* wsrc[16] = { aLL.wq, aLL.wk, aLL.wv, aLL.wo,
                            aLH.wq, aLH.wk, aLH.wv, aLH.wo,
                            aX.wq,  aX.wk,  aX.wv,  aX.wo,
                            mLL_w1, mLL_w2, mLH_w1, mLH_w2 };
  const int wK[16] = { 768,768,768,768, 768,768,768,768, 768,768,768,768,
                       768, D_FF, 768, D_FF };
  const int wN[16] = { 768,768,768,768, 768,768,768,768, 768,768,768,768,
                       D_FF, 768, D_FF, 768 };
  bf16_t* wbf[16];
  for (int i = 0; i < 16; ++i)
    wbf[i] = (bf16_t*)alloc((size_t)wK[i] * wN[i] * 2);
  dim3 tblk(32, 8);
  for (int i = 0; i < 16; ++i) {
    dim3 g(wK[i] / 32, wN[i] / 32);
    conv_t_w_kernel<<<g, tblk, 0, stream>>>(wsrc[i], wbf[i], wK[i], wN[i]);
  }

  dim3 tgrid(SEQ_L / 32, D_MODEL / 32, BATCH);
  transpose_in_kernel<<<tgrid, tblk, 0, stream>>>(LL, seqLL_F, seqLL_B);
  transpose_in_kernel<<<tgrid, tblk, 0, stream>>>(LH, seqLH_F, seqLH_B);
  energy_kernel<<<ROWS / 8, 256, 0, stream>>>(seqLH_F, energy);

  // GEMM dispatch helpers (template instantiations; N,K multiples of 128/32)
  auto gemmB = [&](const bf16_t* A, const bf16_t* Wt, const float* bias,
                   bf16_t* Cb, int M, int N, int K) {
    gemm_bf16_kernel<0, false, true, false>
        <<<dim3(M / 128, N / 128), 256, 0, stream>>>(A, Wt, bias, nullptr, Cb,
                                                     nullptr, M, N, K);
  };
  auto gemmVT = [&](const bf16_t* A, const bf16_t* Wt, const float* bias,
                    bf16_t* CbT, int M, int N, int K) {
    gemm_bf16_kernel<0, false, false, true>
        <<<dim3(M / 128, N / 128), 256, 0, stream>>>(A, Wt, bias, nullptr,
                                                     nullptr, CbT, M, N, K);
  };
  auto gemmFB = [&](const bf16_t* A, const bf16_t* Wt, const float* bias,
                    float* Cf, bf16_t* Cb, int M, int N, int K) {
    gemm_bf16_kernel<0, true, true, false>
        <<<dim3(M / 128, N / 128), 256, 0, stream>>>(A, Wt, bias, Cf, Cb,
                                                     nullptr, M, N, K);
  };
  auto gemmF = [&](const bf16_t* A, const bf16_t* Wt, const float* bias,
                   float* Cf, int M, int N, int K) {
    gemm_bf16_kernel<0, true, false, false>
        <<<dim3(M / 128, N / 128), 256, 0, stream>>>(A, Wt, bias, Cf, nullptr,
                                                     nullptr, M, N, K);
  };
  auto gemmGeluB = [&](const bf16_t* A, const bf16_t* Wt, const float* bias,
                       bf16_t* Cb, int M, int N, int K) {
    gemm_bf16_kernel<1, false, true, false>
        <<<dim3(M / 128, N / 128), 256, 0, stream>>>(A, Wt, bias, nullptr, Cb,
                                                     nullptr, M, N, K);
  };
  dim3 agrid(SEQ_L / 64, NH, BATCH);

  // ---- LL self-attention ----
  gemmB (seqLL_B, wbf[0], aLL.bq, qB,  ROWS, D_MODEL, D_MODEL);
  gemmB (seqLL_B, wbf[1], aLL.bk, kB,  ROWS, D_MODEL, D_MODEL);
  gemmVT(seqLL_B, wbf[2], aLL.bv, vTB, ROWS, D_MODEL, D_MODEL);
  flash_attn_kernel<false><<<agrid, 128, 0, stream>>>(qB, kB, vTB, nullptr,
                                                      TAU_F, attnB);
  gemmFB(attnB, wbf[3], aLL.bo, outF_LL, outB_LL, ROWS, D_MODEL, D_MODEL);

  // ---- LH self-attention (frequency-sparse) ----
  gemmB (seqLH_B, wbf[4], aLH.bq, qB,  ROWS, D_MODEL, D_MODEL);
  gemmB (seqLH_B, wbf[5], aLH.bk, kB,  ROWS, D_MODEL, D_MODEL);
  gemmVT(seqLH_B, wbf[6], aLH.bv, vTB, ROWS, D_MODEL, D_MODEL);
  flash_attn_kernel<true><<<agrid, 128, 0, stream>>>(qB, kB, vTB, energy,
                                                     TAU_F, attnB);
  gemmFB(attnB, wbf[7], aLH.bo, outF_LH, outB_LH, ROWS, D_MODEL, D_MODEL);

  // ---- cross attention: Q from LL_out, K/V from LH_out ----
  gemmB (outB_LL, wbf[8],  aX.bq, qB,  ROWS, D_MODEL, D_MODEL);
  gemmB (outB_LH, wbf[9],  aX.bk, kB,  ROWS, D_MODEL, D_MODEL);
  gemmVT(outB_LH, wbf[10], aX.bv, vTB, ROWS, D_MODEL, D_MODEL);
  flash_attn_kernel<false><<<agrid, 128, 0, stream>>>(qB, kB, vTB, nullptr,
                                                      TAU_F, attnB);
  gemmF (attnB, wbf[11], aX.bo, outF_X, ROWS, D_MODEL, D_MODEL);

  // ---- fuse + layernorm ----
  add_ln_kernel<true><<<ROWS / 8, 256, 0, stream>>>(seqLL_F, outF_LL, outF_X,
                                                    g1, b1n, fusLL_F, fusLL_B);
  add_ln_kernel<false><<<ROWS / 8, 256, 0, stream>>>(seqLH_F, outF_LH, nullptr,
                                                     g2, b2n, fusLH_F, fusLH_B);

  float* outLL = (float*)d_out;
  float* outLH = outLL + NE;

  // ---- MLP LL (+residual, transpose out) ----
  gemmGeluB(fusLL_B, wbf[12], mLL_b1, hiddenB, ROWS, D_FF, D_MODEL);
  gemmF    (hiddenB, wbf[13], mLL_b2, mlpF, ROWS, D_MODEL, D_FF);
  add_transpose_out_kernel<<<tgrid, tblk, 0, stream>>>(mlpF, fusLL_F, outLL);

  // ---- MLP LH (+residual, transpose out) ----
  gemmGeluB(fusLH_B, wbf[14], mLH_b1, hiddenB, ROWS, D_FF, D_MODEL);
  gemmF    (hiddenB, wbf[15], mLH_b2, mlpF, ROWS, D_MODEL, D_FF);
  add_transpose_out_kernel<<<tgrid, tblk, 0, stream>>>(mlpF, fusLH_F, outLH);
}
